// ChaoticLSTM_80479097192689
// MI455X (gfx1250) — matrix operand order repaired and checked by reading, not compiled, not run on hardware
//
#include <hip/hip_runtime.h>
#include <hip/hip_fp16.h>

typedef _Float16 v16h __attribute__((ext_vector_type(16)));
typedef _Float16 h8v  __attribute__((ext_vector_type(8)));
typedef float    v8f  __attribute__((ext_vector_type(8)));

// Problem dimensions
constexpr int BS   = 64;
constexpr int SEQ  = 512;
constexpr int IN   = 128;
constexpr int H    = 256;
constexpr int NWG  = 16;          // workgroups in recurrent kernel (one per WGP)
constexpr int STEPS = SEQ;

// Workspace layout (bytes)
constexpr size_t OFF_BAR = 0;                       // 4B barrier counter
constexpr size_t OFF_H   = 1024;                    // 2 x (64*256) f16 h ping-pong = 64KB
constexpr size_t OFF_B   = OFF_H + 2 * BS * H * 2;  // 66560: packed bias, 16*64 f32
constexpr size_t OFF_WI  = 131072;                  // packed Wi f16: 16*4*4*32*16 halves (256KB)
constexpr size_t OFF_WH  = OFF_WI + 262144;         // packed Wh f16: 16*4*8*32*16 halves (512KB)

// Prep work-item counts
constexpr int N_WH  = 262144;
constexpr int N_WI  = 131072;
constexpr int N_B   = 1024;
constexpr int N_HZ  = 2 * BS * H;                   // 32768 halves to zero
constexpr int N_TOT = N_WH + N_WI + N_B + N_HZ + 1;

static __device__ __forceinline__ v8f wmma_f16(v16h a, v16h b, v8f c) {
  return __builtin_amdgcn_wmma_f32_16x16x32_f16(false, a, false, b, (short)0, c, false, false);
}

static __device__ __forceinline__ float fast_sigmoid(float x) {
  return 1.0f / (1.0f + __expf(-x));
}
static __device__ __forceinline__ float fast_tanh(float x) {
  // tanh(x) = 2*sigmoid(2x) - 1, uses v_exp_f32 transcendental path
  return 2.0f / (1.0f + __expf(-2.0f * x)) - 1.0f;
}

// ---------------------------------------------------------------------------
// Prep: pack Wi/Wh into this file's WMMA B-operand lane layout (f16), pack bias,
// zero h ping-pong buffers and the global barrier counter.
// B-operand layout used here (contiguous k, same as A): lane L holds column n = L%16,
// halves h=0..15 hold K = (L/16)*16 + h within each 32-K tile.
// ---------------------------------------------------------------------------
__global__ void lstm_prep(const float* __restrict__ Wi,
                          const float* __restrict__ Wh,
                          const float* __restrict__ B,
                          unsigned char* __restrict__ ws) {
  _Float16* WiP = (_Float16*)(ws + OFF_WI);
  _Float16* WhP = (_Float16*)(ws + OFF_WH);
  float*    Bp  = (float*)(ws + OFF_B);
  _Float16* hb  = (_Float16*)(ws + OFF_H);
  unsigned* bar = (unsigned*)(ws + OFF_BAR);

  int id = blockIdx.x * blockDim.x + threadIdx.x;
  if (id < N_WH) {
    // e = ((w*4 + nt)*8 + kt)*512 + L*16 + h
    int h  = id & 15;
    int L  = (id >> 4) & 31;
    int kt = (id >> 9) & 7;
    int nt = (id >> 12) & 3;
    int w  = id >> 14;
    int k  = kt * 32 + (L >> 4) * 16 + h;            // 0..255
    int g  = nt * 256 + w * 16 + (L & 15);           // gate column 0..1023
    WhP[id] = (_Float16)Wh[k * 1024 + g];
  } else if (id < N_WH + N_WI) {
    int e  = id - N_WH;                               // e = ((w*4+nt)*4+kt)*512 + L*16 + h
    int h  = e & 15;
    int L  = (e >> 4) & 31;
    int kt = (e >> 9) & 3;
    int nt = (e >> 11) & 3;
    int w  = e >> 13;
    int k  = kt * 32 + (L >> 4) * 16 + h;            // 0..127
    int g  = nt * 256 + w * 16 + (L & 15);
    WiP[e] = (_Float16)Wi[k * 1024 + g];
  } else if (id < N_WH + N_WI + N_B) {
    int e  = id - (N_WH + N_WI);                      // e = w*64 + nt*16 + jl
    int jl = e & 15;
    int nt = (e >> 4) & 3;
    int w  = e >> 6;
    Bp[e] = B[nt * 256 + w * 16 + jl];
  } else if (id < N_WH + N_WI + N_B + N_HZ) {
    hb[id - (N_WH + N_WI + N_B)] = (_Float16)0.0f;
  } else if (id == N_TOT - 1) {
    *bar = 0u;
  }
}

// ---------------------------------------------------------------------------
// Recurrent kernel: 16 WGs x 128 threads (4 waves). WG w owns hidden units
// j in [16w, 16w+16). Wave u owns batch rows [16u, 16u+16).
// Per step: gates[64x64] = x_t[64x128] @ WiP + h[64x256] @ WhP + bias,
// then per-lane LSTM cell update; h broadcast via L2 ping-pong + atomic barrier.
// x rows are cache-prefetched one step ahead; h loads are hoisted to the step
// top and hidden under the Wi WMMA section; f32 output stores are deferred
// past the barrier arrive.
// ---------------------------------------------------------------------------
__global__ __launch_bounds__(128) void lstm_recurrent(const float* __restrict__ x,
                                                      float* __restrict__ out,
                                                      unsigned char* __restrict__ ws) {
  __shared__ __align__(16) _Float16 WiL[8192];    // 16KB: [nt][kt4][lane][16]
  __shared__ __align__(16) _Float16 WhL[16384];   // 32KB: [nt][kt8][lane][16]

  const int w   = blockIdx.x;       // 0..15
  const int tid = threadIdx.x;
  const int u   = tid >> 5;         // wave 0..3
  const int L   = tid & 31;         // lane
  const int jl  = L & 15;
  const int khalf = (L >> 4) * 16;  // A-operand K sub-offset per lane half

  const _Float16* WiP = (const _Float16*)(ws + OFF_WI) + (size_t)w * 8192;
  const _Float16* WhP = (const _Float16*)(ws + OFF_WH) + (size_t)w * 16384;
  const float*    Bp  = (const float*)(ws + OFF_B) + w * 64;
  _Float16*       hb  = (_Float16*)(ws + OFF_H);
  unsigned*       bar = (unsigned*)(ws + OFF_BAR);

  // Stage packed weight slices into LDS (128-bit copies).
  {
    const uint4* s0 = (const uint4*)WiP;  uint4* d0 = (uint4*)WiL;
    for (int i = tid; i < 1024; i += 128) d0[i] = s0[i];
    const uint4* s1 = (const uint4*)WhP;  uint4* d1 = (uint4*)WhL;
    for (int i = tid; i < 2048; i += 128) d1[i] = s1[i];
  }

  float bias[4];
#pragma unroll
  for (int nt = 0; nt < 4; ++nt) bias[nt] = Bp[nt * 16 + jl];
  __syncthreads();

  const int bbase = u * 16;
  const int bm    = bbase + (L & 15);            // A-operand row (batch) for this lane
  const int bc0   = bbase + (L >> 4) * 8;        // C-operand base row for this lane
  const int j     = w * 16 + jl;                 // hidden unit for this lane's C columns
  const float* xrow_base = x + (size_t)bm * SEQ * IN;

  float* outH = out + (size_t)BS * SEQ * H;
  float* outC = outH + (size_t)BS * H;

  v8f cs;                                        // cell state, lives in VGPRs
#pragma unroll
  for (int v = 0; v < 8; ++v) cs[v] = 0.0f;

  // B-operand fetch from LDS (two 16B ds loads -> one v16h).
  auto ldsB = [&](const _Float16* base, int idx) -> v16h {
    const h8v* p = (const h8v*)(base + (size_t)idx * 16);
    return __builtin_shufflevector(p[0], p[1],
             0,1,2,3,4,5,6,7,8,9,10,11,12,13,14,15);
  };

  // Warm caches for t = 0.
  {
    const char* nx = (const char*)xrow_base + (L >> 4) * 256;
    __builtin_prefetch(nx, 0, 3);
    __builtin_prefetch(nx + 128, 0, 3);
  }

  for (int t = 0; t < STEPS; ++t) {
    const _Float16* hin  = hb + (size_t)(t & 1) * (BS * H);
    _Float16*       hout = hb + (size_t)((t + 1) & 1) * (BS * H);

    // ---- x A-tiles for this step (prefetch-warmed; issued before h loads
    //      so the ax loadcnt wait does not cover the h loads) ----
    v16h ax[4];
    {
      const float* xr = xrow_base + (size_t)t * IN;
#pragma unroll
      for (int kt = 0; kt < 4; ++kt) {
        const int k0 = kt * 32 + khalf;
        float4 f0 = *(const float4*)(xr + k0);
        float4 f1 = *(const float4*)(xr + k0 + 4);
        float4 f2 = *(const float4*)(xr + k0 + 8);
        float4 f3 = *(const float4*)(xr + k0 + 12);
        v16h a;
        a[0]=(_Float16)f0.x; a[1]=(_Float16)f0.y; a[2]=(_Float16)f0.z; a[3]=(_Float16)f0.w;
        a[4]=(_Float16)f1.x; a[5]=(_Float16)f1.y; a[6]=(_Float16)f1.z; a[7]=(_Float16)f1.w;
        a[8]=(_Float16)f2.x; a[9]=(_Float16)f2.y; a[10]=(_Float16)f2.z; a[11]=(_Float16)f2.w;
        a[12]=(_Float16)f3.x; a[13]=(_Float16)f3.y; a[14]=(_Float16)f3.z; a[15]=(_Float16)f3.w;
        ax[kt] = a;
      }
    }

    // ---- issue ALL h A-operand loads (hidden under the Wi WMMA section) ----
    const _Float16* hrow = hin + bm * H;
    h8v hlo[8], hhi[8];
#pragma unroll
    for (int kt = 0; kt < 8; ++kt) {
      const int k0 = kt * 32 + khalf;
      hlo[kt] = *(const h8v*)(hrow + k0);
      hhi[kt] = *(const h8v*)(hrow + k0 + 8);
    }

    // ---- prefetch next step's x row (no VGPR cost) ----
    if (t + 1 < STEPS) {
      const char* nx = (const char*)(xrow_base + (size_t)(t + 1) * IN) + (L >> 4) * 256;
      __builtin_prefetch(nx, 0, 3);
      __builtin_prefetch(nx + 128, 0, 3);
    }

    v8f acc[4];
#pragma unroll
    for (int nt = 0; nt < 4; ++nt)
#pragma unroll
      for (int v = 0; v < 8; ++v) acc[nt][v] = bias[nt];

    // ---- input projection: x_t[64x128] @ WiP (K = 128 -> 4 k-tiles) ----
#pragma unroll
    for (int kt = 0; kt < 4; ++kt) {
      // batch the 4 gate-tile B loads so dscnt waits overlap the WMMAs
      v16h b0 = ldsB(WiL, (0 * 4 + kt) * 32 + L);
      v16h b1 = ldsB(WiL, (1 * 4 + kt) * 32 + L);
      v16h b2 = ldsB(WiL, (2 * 4 + kt) * 32 + L);
      v16h b3 = ldsB(WiL, (3 * 4 + kt) * 32 + L);
      acc[0] = wmma_f16(ax[kt], b0, acc[0]);
      acc[1] = wmma_f16(ax[kt], b1, acc[1]);
      acc[2] = wmma_f16(ax[kt], b2, acc[2]);
      acc[3] = wmma_f16(ax[kt], b3, acc[3]);
    }

    // ---- recurrent projection: h[64x256] @ WhP (K = 256 -> 8 k-tiles) ----
#pragma unroll
    for (int kt = 0; kt < 8; ++kt) {
      v16h a = __builtin_shufflevector(hlo[kt], hhi[kt],
                 0,1,2,3,4,5,6,7,8,9,10,11,12,13,14,15);
      v16h b0 = ldsB(WhL, (0 * 8 + kt) * 32 + L);
      v16h b1 = ldsB(WhL, (1 * 8 + kt) * 32 + L);
      v16h b2 = ldsB(WhL, (2 * 8 + kt) * 32 + L);
      v16h b3 = ldsB(WhL, (3 * 8 + kt) * 32 + L);
      acc[0] = wmma_f16(a, b0, acc[0]);
      acc[1] = wmma_f16(a, b1, acc[1]);
      acc[2] = wmma_f16(a, b2, acc[2]);
      acc[3] = wmma_f16(a, b3, acc[3]);
    }

    // ---- gates + cell update (all four gates share this lane's slots) ----
    float hv[8];
#pragma unroll
    for (int v = 0; v < 8; ++v) {
      float ig = fast_sigmoid(acc[0][v]);
      float fg = fast_sigmoid(acc[1][v]);
      float gg = fast_tanh(acc[2][v]);
      float og = fast_sigmoid(acc[3][v]);
      float cv = fg * cs[v] + ig * gg;
      cs[v] = cv;
      hv[v] = og * fast_tanh(cv);
      // f16 h broadcast store: this is all other WGPs need from us
      hout[(bc0 + v) * H + j] = (_Float16)hv[v];
    }

    if (t != STEPS - 1) {
      // ---- make h visible, then arrive; defer f32 output stores to the
      //      spin window so they are off the inter-WG critical path ----
      __threadfence();
      __syncthreads();
      if (tid == 0) {
        __hip_atomic_fetch_add(bar, 1u, __ATOMIC_RELEASE, __HIP_MEMORY_SCOPE_AGENT);
      }
#pragma unroll
      for (int v = 0; v < 8; ++v) {
        out[((size_t)(bc0 + v) * SEQ + t) * H + j] = hv[v];
      }
      if (tid == 0) {
        const unsigned target = 16u * (unsigned)(t + 1);
        while (__hip_atomic_load(bar, __ATOMIC_ACQUIRE, __HIP_MEMORY_SCOPE_AGENT) < target) {
        }
      }
      __syncthreads();
    } else {
#pragma unroll
      for (int v = 0; v < 8; ++v) {
        const int b = bc0 + v;
        out[((size_t)b * SEQ + t) * H + j] = hv[v];
        outH[b * H + j] = hv[v];
        outC[b * H + j] = cs[v];
      }
    }
  }
}

extern "C" void kernel_launch(void* const* d_in, const int* in_sizes, int n_in,
                              void* d_out, int out_size, void* d_ws, size_t ws_size,
                              hipStream_t stream) {
  (void)in_sizes; (void)n_in; (void)out_size; (void)ws_size;
  const float* x  = (const float*)d_in[0];
  const float* Wi = (const float*)d_in[1];
  const float* Wh = (const float*)d_in[2];
  const float* B  = (const float*)d_in[3];
  float* out = (float*)d_out;
  unsigned char* ws = (unsigned char*)d_ws;

  const int prep_blocks = (N_TOT + 255) / 256;
  lstm_prep<<<prep_blocks, 256, 0, stream>>>(Wi, Wh, B, ws);
  lstm_recurrent<<<NWG, 128, 0, stream>>>(x, out, ws);
}
